// GKN_FNO_4080218931796
// MI455X (gfx1250) — matrix operand order repaired and checked by reading, not compiled, not run on hardware
//
#include <hip/hip_runtime.h>
#include <hip/hip_bf16.h>
#include <math.h>

typedef __attribute__((ext_vector_type(16))) _Float16 v16h;
typedef __attribute__((ext_vector_type(8)))  float    v8f;

#define NPIX 65536     // 256*256
#define HW   256
#define NCH  32
#define EEDGE 1048576

union V16H { _Float16 h[16]; v16h v; };
union V8F  { float    f[8];  v8f  v; };

// Branch-free gelu (tanh approximation, matches jax.nn.gelu approximate=True).
// tanh(t) = 1 - 2/(exp(2t)+1); exp via single v_exp_f32 (base-2), rcp via
// v_rcp_f32. No libm range-reduction branches -> no EXEC churn between WMMAs.
__device__ __forceinline__ float gelu_f(float x) {
    float t  = 0.7978845608028654f * (x + 0.044715f * x * x * x);
    float e2 = __builtin_amdgcn_exp2f(2.885390081777927f * t);   // exp(2t)
    float th = 1.0f - 2.0f * __builtin_amdgcn_rcpf(e2 + 1.0f);
    return 0.5f * x * (1.0f + th);
}

// cos(2*pi*r), r in revolutions: native V_COS_F32 parameterization.
__device__ __forceinline__ float cos_rev(float r) {
    return __builtin_amdgcn_cosf(r);
}

// ---------------------------------------------------------------------------
// counter scatter: counter[ind1[e]] += 1
// ---------------------------------------------------------------------------
__global__ __launch_bounds__(256) void k_counter(const int* __restrict__ ind1,
                                                 float* __restrict__ counter) {
    int e = blockIdx.x * 256 + threadIdx.x;
    atomicAdd(&counter[ind1[e]], 1.0f);
}

// ---------------------------------------------------------------------------
// Edge MLP with WMMA f16 (f32 accumulate). 8 waves/block, 16 edges per wave
// per iteration, 8 iterations -> 128 edges/wave, 1024 edges/block.
// ---------------------------------------------------------------------------
__global__ __launch_bounds__(256)
void k_edge_mlp(const float* __restrict__ u, const float* __restrict__ x0,
                const float* __restrict__ x1,
                const int* __restrict__ ind0, const int* __restrict__ ind1,
                const float* __restrict__ g0w, const float* __restrict__ g0b,
                const float* __restrict__ g1w, const float* __restrict__ g1b,
                const float* __restrict__ g2w, const float* __restrict__ g2b,
                const float* __restrict__ g3w, const float* __restrict__ g3b,
                float* __restrict__ acc) {
    __shared__ __align__(32) _Float16 sW0[4 * 32 * 16];   // W0 A-tiles (K padded 10->32)
    __shared__ __align__(32) _Float16 sW1[8 * 32 * 16];   // W1 A-tiles (mt*2+kc)
    __shared__ __align__(32) _Float16 sW2[8 * 32 * 16];   // W2 A-tiles
    __shared__ float sB0[64], sB1[64], sB2[64], sW3[64];
    __shared__ float sB3;
    __shared__ __align__(32) _Float16 sAct[8][16 * 64];   // per-wave activations [col][k]

    const int tid = threadIdx.x;

    // Pack W0 (64x10, pad K to 32) into A-tile layout.
    for (int idx = tid; idx < 4 * 32 * 16; idx += 256) {
        int e = idx & 15, l = (idx >> 4) & 31, mt = idx >> 9;
        int m = mt * 16 + (l & 15);
        int k = ((l >> 4) << 4) + e;
        sW0[idx] = (_Float16)((k < 10) ? g0w[m * 10 + k] : 0.0f);
    }
    // Pack W1 / W2 (64x64) into A-tile layout: tile t = mt*2 + kc.
    for (int idx = tid; idx < 8 * 32 * 16; idx += 256) {
        int e = idx & 15, l = (idx >> 4) & 31, t = idx >> 9;
        int mt = t >> 1, kc = t & 1;
        int m = mt * 16 + (l & 15);
        int k = kc * 32 + ((l >> 4) << 4) + e;
        sW1[idx] = (_Float16)g1w[m * 64 + k];
        sW2[idx] = (_Float16)g2w[m * 64 + k];
    }
    for (int i = tid; i < 64; i += 256) {
        sB0[i] = g0b[i]; sB1[i] = g1b[i]; sB2[i] = g2b[i]; sW3[i] = g3w[i];
    }
    if (tid == 0) sB3 = g3b[0];
    __syncthreads();

    const int wave = tid >> 5, lane = tid & 31;
    const int hi = lane >> 4, col = lane & 15;
    _Float16* act = &sAct[wave][0];

    for (int it = 0; it < 8; ++it) {
        int tile = blockIdx.x * 64 + wave * 8 + it;
        int e = tile * 16 + col;
        int i0 = ind0[e], i1 = ind1[e];
        __builtin_prefetch(ind0 + e + 2048, 0, 0);   // gfx1250 global_prefetch_b8

        // ---- build B0 (10 features, K padded to 32): B layout k = hi*16 + e
        V16H b0;
#pragma unroll
        for (int q = 0; q < 16; ++q) b0.h[q] = (_Float16)0.0f;
        if (hi == 0) {
            float f0 = x0[i0], f1 = x0[NPIX + i0], f2 = u[i0];
            b0.h[0] = (_Float16)f0;            b0.h[1] = (_Float16)f1;
            b0.h[2] = (_Float16)f2;
            b0.h[3] = (_Float16)x0[i1];        b0.h[4] = (_Float16)x0[NPIX + i1];
            b0.h[5] = (_Float16)u[i1];
            b0.h[6] = (_Float16)f0;            b0.h[7] = (_Float16)f1;
            b0.h[8] = (_Float16)x1[i0];        b0.h[9] = (_Float16)x1[NPIX + i0];
        }

        // ---- layer 0: h0 = W0 @ feats + b0  (4 M-tiles, one K step)
        V8F h[4];
#pragma unroll
        for (int mt = 0; mt < 4; ++mt) {
            V8F c;
#pragma unroll
            for (int r = 0; r < 8; ++r) c.f[r] = sB0[mt * 16 + hi * 8 + r];
            v16h a = *(const v16h*)&sW0[(mt * 32 + lane) * 16];
            h[mt].v = __builtin_amdgcn_wmma_f32_16x16x32_f16(
                false, a, false, b0.v, (short)0, c.v, false, false);
        }
#pragma unroll
        for (int mt = 0; mt < 4; ++mt)
#pragma unroll
            for (int r = 0; r < 8; ++r)
                act[col * 64 + mt * 16 + hi * 8 + r] = (_Float16)gelu_f(h[mt].f[r]);

        // ---- layer 1: 64x64, 2 K-chunks chained through C
        v16h bb0 = *(const v16h*)&act[col * 64 + hi * 16];
        v16h bb1 = *(const v16h*)&act[col * 64 + 32 + hi * 16];
#pragma unroll
        for (int mt = 0; mt < 4; ++mt) {
            V8F c;
#pragma unroll
            for (int r = 0; r < 8; ++r) c.f[r] = sB1[mt * 16 + hi * 8 + r];
            v16h a0 = *(const v16h*)&sW1[((mt * 2 + 0) * 32 + lane) * 16];
            v16h a1 = *(const v16h*)&sW1[((mt * 2 + 1) * 32 + lane) * 16];
            v8f d = __builtin_amdgcn_wmma_f32_16x16x32_f16(
                false, a0, false, bb0, (short)0, c.v, false, false);
            d = __builtin_amdgcn_wmma_f32_16x16x32_f16(
                false, a1, false, bb1, (short)0, d, false, false);
            h[mt].v = d;
        }
#pragma unroll
        for (int mt = 0; mt < 4; ++mt)
#pragma unroll
            for (int r = 0; r < 8; ++r)
                act[col * 64 + mt * 16 + hi * 8 + r] = (_Float16)gelu_f(h[mt].f[r]);

        // ---- layer 2
        bb0 = *(const v16h*)&act[col * 64 + hi * 16];
        bb1 = *(const v16h*)&act[col * 64 + 32 + hi * 16];
#pragma unroll
        for (int mt = 0; mt < 4; ++mt) {
            V8F c;
#pragma unroll
            for (int r = 0; r < 8; ++r) c.f[r] = sB2[mt * 16 + hi * 8 + r];
            v16h a0 = *(const v16h*)&sW2[((mt * 2 + 0) * 32 + lane) * 16];
            v16h a1 = *(const v16h*)&sW2[((mt * 2 + 1) * 32 + lane) * 16];
            v8f d = __builtin_amdgcn_wmma_f32_16x16x32_f16(
                false, a0, false, bb0, (short)0, c.v, false, false);
            d = __builtin_amdgcn_wmma_f32_16x16x32_f16(
                false, a1, false, bb1, (short)0, d, false, false);
            h[mt].v = d;
        }
#pragma unroll
        for (int mt = 0; mt < 4; ++mt)
#pragma unroll
            for (int r = 0; r < 8; ++r)
                act[col * 64 + mt * 16 + hi * 8 + r] = (_Float16)gelu_f(h[mt].f[r]);

        // ---- layer 3 (1x64 head) + scatter: lanes 0..15 own one edge each
        if (hi == 0) {
            float w = sB3;
#pragma unroll 8
            for (int k = 0; k < 64; ++k) w += sW3[k] * (float)act[col * 64 + k];
            atomicAdd(&acc[i1],            w * x0[i0]);
            atomicAdd(&acc[NPIX + i1],     w * x0[NPIX + i0]);
            atomicAdd(&acc[2 * NPIX + i1], w * u[i0]);
        }
    }
}

// ---------------------------------------------------------------------------
// v3 = acc / (counter==0 ? 1 : counter)
// ---------------------------------------------------------------------------
__global__ __launch_bounds__(256)
void k_gkn_norm(const float* __restrict__ acc, const float* __restrict__ counter,
                float* __restrict__ v3) {
    int idx = blockIdx.x * 256 + threadIdx.x;    // 3*NPIX threads
    int n = idx % NPIX;
    float c = counter[n];
    c = (c == 0.0f) ? 1.0f : c;
    v3[idx] = acc[idx] * __builtin_amdgcn_rcpf(c);
}

// ---------------------------------------------------------------------------
// encoder conv1x1: 3 -> 32 (K too small for WMMA; pure VALU)
// ---------------------------------------------------------------------------
__global__ __launch_bounds__(256)
void k_enc(const float* __restrict__ v3, const float* __restrict__ w,
           const float* __restrict__ b, float* __restrict__ v32) {
    int p = blockIdx.x * 256 + threadIdx.x;
    float a0 = v3[p], a1 = v3[NPIX + p], a2 = v3[2 * NPIX + p];
#pragma unroll 8
    for (int o = 0; o < 32; ++o)
        v32[o * NPIX + p] = b[o] + w[o * 3] * a0 + w[o * 3 + 1] * a1 + w[o * 3 + 2] * a2;
}

// ---------------------------------------------------------------------------
// Truncated DFT, x-stage: s[ch][kx][y] = sum_x v[ch][y][x] e^{-2pi i kx x/256}
// ---------------------------------------------------------------------------
__global__ __launch_bounds__(256)
void k_dft_x(const float* __restrict__ v, float* __restrict__ sbuf) {
    __shared__ float rows[8 * 256];
    __shared__ float ct[256];
    int b = blockIdx.x, ch = b >> 5, yg = b & 31, t = threadIdx.x;
    ct[t] = cos_rev((float)t * 0.00390625f);          // cos(2*pi*t/256)
    for (int idx = t; idx < 2048; idx += 256)
        rows[idx] = v[ch * NPIX + (yg * 8 + (idx >> 8)) * HW + (idx & 255)];
    __syncthreads();
    int yy = t >> 5, o = t & 31, kx = o >> 1, part = o & 1;
    const float* row = &rows[yy * 256];
    float a = 0.0f;
    if (part == 0) {
        for (int x = 0; x < 256; ++x) a += row[x] * ct[(kx * x) & 255];
    } else {
        for (int x = 0; x < 256; ++x) a -= row[x] * ct[((kx * x) + 192) & 255];
    }
    sbuf[((ch * 16 + kx) * 256 + yg * 8 + yy) * 2 + part] = a;
}

// ---------------------------------------------------------------------------
// Truncated DFT, y-stage (rows ky = -8..7) with 1/(H*W) forward norm.
// ---------------------------------------------------------------------------
__global__ __launch_bounds__(256)
void k_dft_y(const float* __restrict__ sbuf, float* __restrict__ coeff) {
    __shared__ float ct[256];
    int ch = blockIdx.x, t = threadIdx.x;
    ct[t] = cos_rev((float)t * 0.00390625f);
    __syncthreads();
    int j = t >> 4, kx = t & 15;
    int f = (j + 248) & 255;                      // ky = j - 8 mod 256
    const float* s = &sbuf[(ch * 16 + kx) * 512];
    float re = 0.0f, im = 0.0f;
    for (int y = 0; y < 256; ++y) {
        float sr = s[y * 2], si = s[y * 2 + 1];
        int idx = (f * y) & 255;
        float c = ct[idx], sn = ct[(idx + 192) & 255];
        re += sr * c + si * sn;
        im += si * c - sr * sn;
    }
    coeff[((ch * 16 + j) * 16 + kx) * 2 + 0] = re * 1.52587890625e-05f;
    coeff[((ch * 16 + j) * 16 + kx) * 2 + 1] = im * 1.52587890625e-05f;
}

// ---------------------------------------------------------------------------
// spectral channel mix: spec[o][j][kx] = sum_i A[o][i][j][kx] * coeff[i][j][kx]
// ---------------------------------------------------------------------------
__global__ __launch_bounds__(32)
void k_mix(const float* __restrict__ Ar, const float* __restrict__ Ai,
           const float* __restrict__ coeff, float* __restrict__ spec) {
    __shared__ float cr[32], ci[32];
    int m = blockIdx.x, j = m >> 4, kx = m & 15, t = threadIdx.x;
    cr[t] = coeff[((t * 16 + j) * 16 + kx) * 2 + 0];
    ci[t] = coeff[((t * 16 + j) * 16 + kx) * 2 + 1];
    __syncthreads();
    float re = 0.0f, im = 0.0f;
    for (int i = 0; i < 32; ++i) {
        int idx = ((t * 32 + i) * 16 + j) * 16 + kx;
        float ar = Ar[idx], ai = Ai[idx];
        re += ar * cr[i] - ai * ci[i];
        im += ar * ci[i] + ai * cr[i];
    }
    spec[((t * 16 + j) * 16 + kx) * 2 + 0] = re;
    spec[((t * 16 + j) * 16 + kx) * 2 + 1] = im;
}

// ---------------------------------------------------------------------------
// inverse DFT, y-stage: g[ch][y][kx] = sum_j spec[ch][j][kx] e^{+2pi i f_j y/256}
// ---------------------------------------------------------------------------
__global__ __launch_bounds__(256)
void k_idft_y(const float* __restrict__ spec, float* __restrict__ gbuf) {
    __shared__ float sp[512];
    __shared__ float ct[256];
    int b = blockIdx.x, ch = b >> 5, yg = b & 31, t = threadIdx.x;
    ct[t] = cos_rev((float)t * 0.00390625f);
    for (int idx = t; idx < 512; idx += 256) sp[idx] = spec[ch * 512 + idx];
    __syncthreads();
    int yy = t >> 5, o = t & 31, kx = o >> 1, part = o & 1;
    int y = yg * 8 + yy;
    float a = 0.0f;
    for (int j = 0; j < 16; ++j) {
        float pr = sp[(j * 16 + kx) * 2], pi = sp[(j * 16 + kx) * 2 + 1];
        int f = (j + 248) & 255;
        int idx = (f * y) & 255;
        float c = ct[idx], sn = ct[(idx + 192) & 255];
        a += (part == 0) ? (pr * c - pi * sn) : (pr * sn + pi * c);
    }
    gbuf[((ch * 256 + y) * 16 + kx) * 2 + part] = a;
}

// ---------------------------------------------------------------------------
// inverse DFT, x-stage (real output, Hermitian weights on kx>0)
// ---------------------------------------------------------------------------
__global__ __launch_bounds__(256)
void k_idft_x(const float* __restrict__ gbuf, float* __restrict__ hout) {
    __shared__ float g[32];
    __shared__ float ct[256];
    int b = blockIdx.x, ch = b >> 8, y = b & 255, x = threadIdx.x;
    ct[x] = cos_rev((float)x * 0.00390625f);
    if (x < 32) g[x] = gbuf[(ch * 256 + y) * 32 + x];
    __syncthreads();
    float val = g[0];                                 // Re(g[kx=0])
#pragma unroll
    for (int kx = 1; kx < 16; ++kx) {
        int idx = (kx * x) & 255;
        val += 2.0f * (g[kx * 2] * ct[idx] - g[kx * 2 + 1] * ct[(idx + 192) & 255]);
    }
    hout[ch * NPIX + y * HW + x] = val;
}

// ---------------------------------------------------------------------------
// WMMA conv block: v += gelu(W2 @ gelu(W1 @ h + b1) + b2); 8 waves, 16 px/wave
// ---------------------------------------------------------------------------
__global__ __launch_bounds__(256)
void k_conv_block(const float* __restrict__ hin, float* __restrict__ v,
                  const float* __restrict__ w1, const float* __restrict__ b1,
                  const float* __restrict__ w2, const float* __restrict__ b2) {
    __shared__ __align__(32) _Float16 sA1[2 * 32 * 16];
    __shared__ __align__(32) _Float16 sA2[2 * 32 * 16];
    __shared__ float sBias1[32], sBias2[32];
    __shared__ __align__(32) _Float16 sAct[8][16 * 32];
    int tid = threadIdx.x;
    for (int idx = tid; idx < 1024; idx += 256) {
        int e = idx & 15, l = (idx >> 4) & 31, mt = idx >> 9;
        int m = mt * 16 + (l & 15);
        int k = ((l >> 4) << 4) + e;
        sA1[idx] = (_Float16)w1[m * 32 + k];
        sA2[idx] = (_Float16)w2[m * 32 + k];
    }
    if (tid < 32) { sBias1[tid] = b1[tid]; sBias2[tid] = b2[tid]; }
    __syncthreads();

    int wave = tid >> 5, lane = tid & 31;
    int hi = lane >> 4, col = lane & 15;
    int p = (blockIdx.x * 8 + wave) * 16 + col;
    _Float16* act = &sAct[wave][0];

    // B from spectral output: k = hi*16 + e
    V16H bmat;
#pragma unroll
    for (int e = 0; e < 16; ++e)
        bmat.h[e] = (_Float16)hin[(hi * 16 + e) * NPIX + p];

    // conv1 + gelu
#pragma unroll
    for (int mt = 0; mt < 2; ++mt) {
        V8F c;
#pragma unroll
        for (int r = 0; r < 8; ++r) c.f[r] = sBias1[mt * 16 + hi * 8 + r];
        v16h a = *(const v16h*)&sA1[(mt * 32 + lane) * 16];
        V8F d;
        d.v = __builtin_amdgcn_wmma_f32_16x16x32_f16(
            false, a, false, bmat.v, (short)0, c.v, false, false);
#pragma unroll
        for (int r = 0; r < 8; ++r)
            act[col * 32 + mt * 16 + hi * 8 + r] = (_Float16)gelu_f(d.f[r]);
    }
    // conv2 + gelu + residual
    v16h bb = *(const v16h*)&act[col * 32 + hi * 16];
#pragma unroll
    for (int mt = 0; mt < 2; ++mt) {
        V8F c;
#pragma unroll
        for (int r = 0; r < 8; ++r) c.f[r] = sBias2[mt * 16 + hi * 8 + r];
        v16h a = *(const v16h*)&sA2[(mt * 32 + lane) * 16];
        V8F d;
        d.v = __builtin_amdgcn_wmma_f32_16x16x32_f16(
            false, a, false, bb, (short)0, c.v, false, false);
#pragma unroll
        for (int r = 0; r < 8; ++r) {
            int m = mt * 16 + hi * 8 + r;
            v[m * NPIX + p] += gelu_f(d.f[r]);
        }
    }
}

// ---------------------------------------------------------------------------
// decoder conv1x1: 32 -> 1
// ---------------------------------------------------------------------------
__global__ __launch_bounds__(256)
void k_dec(const float* __restrict__ v32, const float* __restrict__ w,
           const float* __restrict__ b, float* __restrict__ out) {
    int p = blockIdx.x * 256 + threadIdx.x;
    float a = b[0];
#pragma unroll 8
    for (int c = 0; c < 32; ++c) a += w[c] * v32[c * NPIX + p];
    out[p] = a;
}

// ---------------------------------------------------------------------------
extern "C" void kernel_launch(void* const* d_in, const int* in_sizes, int n_in,
                              void* d_out, int out_size, void* d_ws, size_t ws_size,
                              hipStream_t stream) {
    const float* u     = (const float*)d_in[0];
    const float* x0    = (const float*)d_in[1];
    const float* x1    = (const float*)d_in[2];
    const float* enc_w = (const float*)d_in[3];
    const float* enc_b = (const float*)d_in[4];
    const float* dec_w = (const float*)d_in[5];
    const float* dec_b = (const float*)d_in[6];
    const float* c1_w  = (const float*)d_in[7];
    const float* c1_b  = (const float*)d_in[8];
    const float* c2_w  = (const float*)d_in[9];
    const float* c2_b  = (const float*)d_in[10];
    const float* A_r   = (const float*)d_in[11];
    const float* A_i   = (const float*)d_in[12];
    const float* g0w   = (const float*)d_in[13];
    const float* g0b   = (const float*)d_in[14];
    const float* g1w   = (const float*)d_in[15];
    const float* g1b   = (const float*)d_in[16];
    const float* g2w   = (const float*)d_in[17];
    const float* g2b   = (const float*)d_in[18];
    const float* g3w   = (const float*)d_in[19];
    const float* g3b   = (const float*)d_in[20];
    const int*   ind0  = (const int*)d_in[21];
    const int*   ind1  = (const int*)d_in[22];
    float* out = (float*)d_out;

    float* ws = (float*)d_ws;
    float* acc     = ws;                        // 3*NPIX
    float* counter = ws + 3 * NPIX;             // NPIX
    float* v3      = ws + 4 * NPIX;             // 3*NPIX
    float* v32     = ws + 7 * NPIX;             // 32*NPIX
    float* hbuf    = ws + 39 * NPIX;            // 32*NPIX
    float* sbuf    = ws + 71 * NPIX;            // 32*16*256*2 = 4*NPIX
    float* coeff   = ws + 75 * NPIX;            // 16384
    float* spec    = coeff + 16384;             // 16384
    float* gbuf    = spec + 16384;              // 4*NPIX

    // zero scatter accumulators + counter
    hipMemsetAsync(d_ws, 0, 4 * NPIX * sizeof(float), stream);

    k_counter<<<EEDGE / 256, 256, 0, stream>>>(ind1, counter);
    k_edge_mlp<<<1024, 256, 0, stream>>>(u, x0, x1, ind0, ind1,
                                         g0w, g0b, g1w, g1b, g2w, g2b, g3w, g3b, acc);
    k_gkn_norm<<<3 * NPIX / 256, 256, 0, stream>>>(acc, counter, v3);
    k_enc<<<NPIX / 256, 256, 0, stream>>>(v3, enc_w, enc_b, v32);

    for (int layer = 0; layer < 4; ++layer) {
        const float* Ar = A_r + (size_t)layer * 32 * 32 * 16 * 16;
        const float* Ai = A_i + (size_t)layer * 32 * 32 * 16 * 16;
        k_dft_x<<<1024, 256, 0, stream>>>(v32, sbuf);
        k_dft_y<<<32, 256, 0, stream>>>(sbuf, coeff);
        k_mix<<<256, 32, 0, stream>>>(Ar, Ai, coeff, spec);
        k_idft_y<<<1024, 256, 0, stream>>>(spec, gbuf);
        k_idft_x<<<32 * 256, 256, 0, stream>>>(gbuf, hbuf);
        k_conv_block<<<512, 256, 0, stream>>>(hbuf, v32,
                                              c1_w + layer * 1024, c1_b + layer * 32,
                                              c2_w + layer * 1024, c2_b + layer * 32);
    }
    k_dec<<<NPIX / 256, 256, 0, stream>>>(v32, dec_w, dec_b, out);
}
